// Conv3x3Norm_17867063951539
// MI455X (gfx1250) — matrix operand order repaired and checked by reading, not compiled, not run on hardware
//
#include <hip/hip_runtime.h>
#include <hip/hip_bf16.h>

typedef __attribute__((ext_vector_type(16))) _Float16     v16h;
typedef __attribute__((ext_vector_type(8)))  float        v8f;
typedef __attribute__((ext_vector_type(4)))  unsigned int v4u;

#define B_    4
#define CIN_  256
#define COUT_ 256
#define H_    64
#define W_    64
#define HW_   4096
#define KTAP_ 9
#define NT_   32      // pixel (N) tile per workgroup
#define KB_   32      // K block = channels per WMMA step
#define ASTR_ 40      // padded LDS row stride in f16 (32 data + 8 pad)

union FragU { v4u q[2]; v16h h; };

// Repack weight (Cout, Cin, 3, 3) f32 -> (Cout, tap, Cin) f16 for contiguous A staging.
__global__ __launch_bounds__(256) void reorder_weights(const float* __restrict__ w,
                                                       _Float16* __restrict__ wr) {
  const int i  = blockIdx.x * 256 + threadIdx.x;   // i = (o*9 + t)*256 + c
  const int c  = i & 255;
  const int ot = i >> 8;
  const int t  = ot % 9;
  const int o  = ot / 9;
  wr[i] = (_Float16)w[(size_t)o * (CIN_ * 9) + c * 9 + t];
}

// Deformable conv as WMMA GEMM: per WG tile = 256 Cout x 32 pixels, K = 9 taps x 256 ch.
__global__ __launch_bounds__(256) void dcn_wmma(
    const float* __restrict__ input,    // (B, Cin, H, W)
    const float* __restrict__ offset,   // (B, 18, H, W)
    const float* __restrict__ mask,     // (B, 9, H, W)
    const _Float16* __restrict__ wr,    // (Cout, 9, Cin) f16
    const float* __restrict__ bias,     // (Cout)
    float* __restrict__ y)              // (B, Cout, H*W)
{
  const int b    = blockIdx.x;
  const int p0   = blockIdx.y * NT_;
  const int tid  = threadIdx.x;
  const int wave = tid >> 5;
  const int lane = tid & 31;
  const int lhalf = (lane < 16) ? 0 : 1;
  const int l15   = lane & 15;

  __shared__ alignas(16) _Float16 Atile[COUT_][ASTR_];  // [m][k] k-contiguous
  __shared__ alignas(16) _Float16 Btile[NT_][ASTR_];    // [n][k] k-contiguous
  __shared__ int   sIdx[NT_][4];
  __shared__ float sWt[NT_][4];

  const float* xb = input + (size_t)b * CIN_ * HW_;

  v8f acc[4] = {};            // [mi*2 + ni]

  const int cl = tid >> 5;    // channel sub-lane 0..7 (consecutive tids share channel)
  const int pl = tid & 31;    // pixel within tile

  for (int t = 0; t < KTAP_; ++t) {
    // Per-tap bilinear setup: 4 clamped gather indices + (bilinear weight * valid * mask)
    if (tid < NT_) {
      const int p  = p0 + tid;
      const int h  = p >> 6, w = p & (W_ - 1);
      const int ki = t / 3,  kj = t % 3;
      const float offy = offset[((size_t)b * 18 + 2 * t    ) * HW_ + p];
      const float offx = offset[((size_t)b * 18 + 2 * t + 1) * HW_ + p];
      const float mk   = mask  [((size_t)b * 9  + t        ) * HW_ + p];
      const float py = offy + (float)(h + ki - 1);
      const float px = offx + (float)(w + kj - 1);
      const float y0f = floorf(py), x0f = floorf(px);
      const float ty = py - y0f, tx = px - x0f;
      const int y0 = (int)y0f, x0 = (int)x0f;
      const int y1 = y0 + 1,   x1 = x0 + 1;
      const int y0c = y0 < 0 ? 0 : (y0 > H_ - 1 ? H_ - 1 : y0);
      const int y1c = y1 < 0 ? 0 : (y1 > H_ - 1 ? H_ - 1 : y1);
      const int x0c = x0 < 0 ? 0 : (x0 > W_ - 1 ? W_ - 1 : x0);
      const int x1c = x1 < 0 ? 0 : (x1 > W_ - 1 ? W_ - 1 : x1);
      const float vy0 = (y0 >= 0 && y0 < H_) ? 1.f : 0.f;
      const float vy1 = (y1 >= 0 && y1 < H_) ? 1.f : 0.f;
      const float vx0 = (x0 >= 0 && x0 < W_) ? 1.f : 0.f;
      const float vx1 = (x1 >= 0 && x1 < W_) ? 1.f : 0.f;
      sIdx[tid][0] = y0c * W_ + x0c;
      sIdx[tid][1] = y0c * W_ + x1c;
      sIdx[tid][2] = y1c * W_ + x0c;
      sIdx[tid][3] = y1c * W_ + x1c;
      sWt[tid][0] = (1.f - ty) * (1.f - tx) * vy0 * vx0 * mk;
      sWt[tid][1] = (1.f - ty) * tx         * vy0 * vx1 * mk;
      sWt[tid][2] = ty         * (1.f - tx) * vy1 * vx0 * mk;
      sWt[tid][3] = ty         * tx         * vy1 * vx1 * mk;
    }
    __syncthreads();

    for (int cb = 0; cb < CIN_ / KB_; ++cb) {
      const int c0 = cb * KB_;

      // Stage A: thread 'tid' copies its Cout row (32 contiguous f16 = 4x b128)
      {
        const v4u* s = (const v4u*)(wr + ((size_t)tid * 9 + t) * CIN_ + c0);
        v4u* d = (v4u*)&Atile[tid][0];
        d[0] = s[0]; d[1] = s[1]; d[2] = s[2]; d[3] = s[3];
      }
      // Stage B: bilinear-sample 32 channels x 32 pixels (4 samples/thread)
      {
        const int   i0 = sIdx[pl][0], i1 = sIdx[pl][1], i2 = sIdx[pl][2], i3 = sIdx[pl][3];
        const float w0 = sWt[pl][0],  w1 = sWt[pl][1],  w2 = sWt[pl][2],  w3 = sWt[pl][3];
#pragma unroll
      for (int i = 0; i < 4; ++i) {
          const int c = c0 + cl + i * 8;
          const float* xc = xb + (size_t)c * HW_;
          const float v = w0 * xc[i0] + w1 * xc[i1] + w2 * xc[i2] + w3 * xc[i3];
          Btile[pl][c - c0] = (_Float16)v;
        }
      }
      __syncthreads();

      // B fragments (contiguous K order: lane -> N, lanes<16 hold K 0..15, lanes>=16 hold K 16..31)
      FragU bf[2];
      const int kbB = lhalf * 16;
#pragma unroll
      for (int ni = 0; ni < 2; ++ni) {
        const v4u* bsrc = (const v4u*)&Btile[ni * 16 + l15][kbB];
        bf[ni].q[0] = bsrc[0];
        bf[ni].q[1] = bsrc[1];
      }
      // A fragments (lane -> M; V0-3 = K kb..kb+7, V4-7 = K kb+8..kb+15, kb = 16*lhalf)
      const int kbA = lhalf * 16;
#pragma unroll
      for (int mi = 0; mi < 2; ++mi) {
        FragU af;
        const _Float16* arow = &Atile[wave * 16 + mi * 128 + l15][0];
        af.q[0] = *(const v4u*)(arow + kbA);
        af.q[1] = *(const v4u*)(arow + 8 + kbA);
#pragma unroll
        for (int ni = 0; ni < 2; ++ni) {
          acc[mi * 2 + ni] = __builtin_amdgcn_wmma_f32_16x16x32_f16(
              false, af.h, false, bf[ni].h, (short)0, acc[mi * 2 + ni], false, false);
        }
      }
      __syncthreads();
    }
  }

  // Store C (ISA C/D layout: VGPR r -> M = r + 8*lhalf, lane -> N) with bias
#pragma unroll
  for (int mi = 0; mi < 2; ++mi) {
#pragma unroll
    for (int ni = 0; ni < 2; ++ni) {
      const v8f a = acc[mi * 2 + ni];
#pragma unroll
      for (int r = 0; r < 8; ++r) {
        const int m = wave * 16 + mi * 128 + r + lhalf * 8;
        const int n = p0 + ni * 16 + l15;
        y[((size_t)b * COUT_ + m) * HW_ + n] = a[r] + bias[m];
      }
    }
  }
}

// Deterministic per-(batch,group) mean / rstd via LDS tree reduction.
__global__ __launch_bounds__(256) void gn_stats(const float* __restrict__ y,
                                                float* __restrict__ stats) {
  const int bg = blockIdx.x;                       // b*16 + g (group = 16 contiguous ch)
  const float* base = y + (size_t)bg * 16 * HW_;
  float s = 0.f, ss = 0.f;
  for (int i = threadIdx.x; i < 16 * HW_; i += 256) {
    const float v = base[i];
    s += v; ss += v * v;
  }
  __shared__ float rs[256], rq[256];
  rs[threadIdx.x] = s; rq[threadIdx.x] = ss;
  __syncthreads();
  for (int st = 128; st > 0; st >>= 1) {
    if (threadIdx.x < st) {
      rs[threadIdx.x] += rs[threadIdx.x + st];
      rq[threadIdx.x] += rq[threadIdx.x + st];
    }
    __syncthreads();
  }
  if (threadIdx.x == 0) {
    const float inv = 1.f / (16.f * HW_);
    const float mu  = rs[0] * inv;
    const float var = rq[0] * inv - mu * mu;
    stats[bg * 2]     = mu;
    stats[bg * 2 + 1] = rsqrtf(var + 1e-5f);
  }
}

// In-place normalize + affine.
__global__ __launch_bounds__(256) void gn_apply(float* __restrict__ y,
                                                const float* __restrict__ stats,
                                                const float* __restrict__ gnw,
                                                const float* __restrict__ gnb) {
  const size_t i = (size_t)blockIdx.x * 256 + threadIdx.x;
  const int c  = (int)((i >> 12) & 255);
  const int bg = (int)(i >> 16);                   // 16ch*4096px = 65536 contiguous / group
  const float mu = stats[bg * 2], rstd = stats[bg * 2 + 1];
  y[i] = (y[i] - mu) * rstd * gnw[c] + gnb[c];
}

extern "C" void kernel_launch(void* const* d_in, const int* in_sizes, int n_in,
                              void* d_out, int out_size, void* d_ws, size_t ws_size,
                              hipStream_t stream) {
  (void)in_sizes; (void)n_in; (void)out_size; (void)ws_size;
  const float* input  = (const float*)d_in[0];
  const float* offset = (const float*)d_in[1];
  const float* maskp  = (const float*)d_in[2];
  const float* weight = (const float*)d_in[3];
  const float* bias   = (const float*)d_in[4];
  const float* gnw    = (const float*)d_in[5];
  const float* gnb    = (const float*)d_in[6];
  float* y = (float*)d_out;

  _Float16* wr = (_Float16*)d_ws;                                  // 256*9*256 f16 = 1.13 MB
  float* stats = (float*)((char*)d_ws + (size_t)COUT_ * 9 * CIN_ * sizeof(_Float16));

  reorder_weights<<<COUT_ * 9, 256, 0, stream>>>(weight, wr);
  dcn_wmma<<<dim3(B_, HW_ / NT_), 256, 0, stream>>>(input, offset, maskp, wr, bias, y);
  gn_stats<<<B_ * 16, 256, 0, stream>>>(y, stats);
  gn_apply<<<(B_ * COUT_ * HW_) / 256, 256, 0, stream>>>(y, stats, gnw, gnb);
}
